// SAUnit_22033182229085
// MI455X (gfx1250) — matrix operand order repaired and checked by reading, not compiled, not run on hardware
//
#include <hip/hip_runtime.h>

// ---------------------------------------------------------------------------
// Types for CDNA5 WMMA (gfx1250, wave32)
// ---------------------------------------------------------------------------
typedef __bf16 v16bf __attribute__((ext_vector_type(16)));
typedef __bf16 v8bf  __attribute__((ext_vector_type(8)));
typedef float  v8f   __attribute__((ext_vector_type(8)));

#define CDIM 1024            // C == H*W == N == 1024
#define BATCH 16
#define MATEL (1048576ull)   // 1024*1024 elements per per-batch matrix

static __device__ __forceinline__ unsigned short f2bf(float f) {
    unsigned u = __builtin_bit_cast(unsigned, f);
    u += 0x7FFFu + ((u >> 16) & 1u);          // round-to-nearest-even
    return (unsigned short)(u >> 16);
}

// ---------------------------------------------------------------------------
// fp32 -> bf16 conversion (vectorized x4)
// ---------------------------------------------------------------------------
__global__ __launch_bounds__(256) void cvt_f32_bf16(const float* __restrict__ s,
                                                    unsigned short* __restrict__ d,
                                                    int n) {
    int i = (blockIdx.x * 256 + threadIdx.x) * 4;
    if (i + 3 < n) {
        float4 v = *(const float4*)(s + i);
        ushort4 o;
        o.x = f2bf(v.x); o.y = f2bf(v.y); o.z = f2bf(v.z); o.w = f2bf(v.w);
        *(ushort4*)(d + i) = o;
    }
}

// ---------------------------------------------------------------------------
// Column softmax over axis c of scores (B, C, C); output bf16 attn[n][d]
// One thread per output column d; coalesced row reads; online max/sum.
// ---------------------------------------------------------------------------
__global__ __launch_bounds__(256) void softmax_col_kernel(const float* __restrict__ S,
                                                          unsigned short* __restrict__ Attn) {
    int d = blockIdx.x * 256 + threadIdx.x;
    size_t base = (size_t)blockIdx.y * MATEL + d;
    float m = -3.4e38f, s = 0.f;
    for (int c = 0; c < CDIM; ++c) {
        float v  = S[base + (size_t)c * CDIM];
        float nm = fmaxf(m, v);
        s = s * __expf(m - nm) + __expf(v - nm);
        m = nm;
    }
    float inv = 1.f / s;
    for (int c = 0; c < CDIM; ++c) {
        float v = S[base + (size_t)c * CDIM];
        Attn[base + (size_t)c * CDIM] = f2bf(__expf(v - m) * inv);
    }
}

// ---------------------------------------------------------------------------
// Tiled bf16 WMMA GEMM: D(M,N) = A(M,K) * B(K,N)  with M=N=K=1024.
//   MODE 0: D bf16 (+ per-row bias if HAS_BIAS)
//   MODE 1: D fp32 (scores)
//   MODE 2: D fp32 with fused bias + LeakyReLU + BatchNorm(eval) + residual x
//   TRANSB: B given as (N,K) row-major (used for f @ g^T)
// Block: 256 threads = 8 waves; tile 128x128, K-step 32, double-buffered LDS.
// Wave (wid): rows (wid>>1)*32, cols (wid&1)*64 -> 2x4 grid of 16x16 WMMA tiles.
// ---------------------------------------------------------------------------
template <int MODE, bool TRANSB, bool HAS_BIAS>
__global__ __launch_bounds__(256) void gemm_wmma(
    const __bf16* __restrict__ A,  long long sA,
    const __bf16* __restrict__ Bm, long long sB,
    void* __restrict__ D,          long long sD,
    const float* __restrict__ bias,
    const float* __restrict__ gamma, const float* __restrict__ beta,
    const float* __restrict__ rmean, const float* __restrict__ rvar,
    const float* __restrict__ xres)
{
    // double-buffered, padded: row stride 80B keeps 16B alignment for b128 ops
    __shared__ __bf16 As[2][128][40];   // [M][K]
    __shared__ __bf16 Bs[2][128][40];   // [N][K] (B stored transposed in LDS)

    const int tid  = threadIdx.x;
    const int lane = tid & 31;
    const int wid  = tid >> 5;
    const int wm   = (wid >> 1) * 32;   // wave row offset in tile
    const int wn   = (wid & 1) * 64;    // wave col offset in tile
    const int m0   = blockIdx.y * 128;
    const int n0   = blockIdx.x * 128;
    const long long b = blockIdx.z;

    A  += b * sA;
    Bm += b * sB;

    v8f acc[2][4] = {};

    const int row = lane & 15;          // fragment row/col within 16
    const int kb  = (lane >> 4) * 8;    // A frag: K half select
    const int kb2 = (lane >> 4) * 16;   // unused; B frag also uses kb

    // per-thread staging coordinates (512 16B-chunks / 256 threads = 2 each)
    const int sm[2] = { (tid + 0) >> 2, (tid + 256) >> 2 };          // contiguous
    const int sk[2] = { ((tid + 0) & 3) * 8, ((tid + 256) & 3) * 8 };
    const int xk[2] = { (tid + 0) & 31, (tid + 256) & 31 };          // scatter
    const int xn[2] = { ((tid + 0) >> 5) * 8, ((tid + 256) >> 5) * 8 };

    // ---- prologue: stage k0 = 0 into buffer 0 ------------------------------
#pragma unroll
    for (int i = 0; i < 2; ++i)
        *(v8bf*)&As[0][sm[i]][sk[i]] =
            *(const v8bf*)&A[(size_t)(m0 + sm[i]) * CDIM + sk[i]];
    if (TRANSB) {
#pragma unroll
        for (int i = 0; i < 2; ++i)
            *(v8bf*)&Bs[0][sm[i]][sk[i]] =
                *(const v8bf*)&Bm[(size_t)(n0 + sm[i]) * CDIM + sk[i]];
    } else {
#pragma unroll
        for (int i = 0; i < 2; ++i) {
            v8bf t = *(const v8bf*)&Bm[(size_t)xk[i] * CDIM + n0 + xn[i]];
#pragma unroll
            for (int j = 0; j < 8; ++j) Bs[0][xn[i] + j][xk[i]] = t[j];
        }
    }

    int cur = 0;
    for (int k0 = 0; k0 < CDIM; k0 += 32) {
        __syncthreads();                       // buffer[cur] ready for all
        const bool havenext = (k0 + 32) < CDIM;
        const int  nxt = cur ^ 1;
        const int  kn  = k0 + 32;

        // ---- issue next tile's global loads early (hide behind WMMAs) ------
        v8bf nA[2], nB[2];
        if (havenext) {
#pragma unroll
            for (int i = 0; i < 2; ++i)
                nA[i] = *(const v8bf*)&A[(size_t)(m0 + sm[i]) * CDIM + kn + sk[i]];
            if (TRANSB) {
#pragma unroll
                for (int i = 0; i < 2; ++i)
                    nB[i] = *(const v8bf*)&Bm[(size_t)(n0 + sm[i]) * CDIM + kn + sk[i]];
            } else {
#pragma unroll
                for (int i = 0; i < 2; ++i)
                    nB[i] = *(const v8bf*)&Bm[(size_t)(kn + xk[i]) * CDIM + n0 + xn[i]];
            }
        }

        // ---- load fragments from buffer[cur] and issue 8 WMMAs -------------
        v16bf af[2], bfr[4];
#pragma unroll
        for (int mi = 0; mi < 2; ++mi) {
            v8bf lo = *(const v8bf*)&As[cur][wm + mi * 16 + row][kb];
            v8bf hi = *(const v8bf*)&As[cur][wm + mi * 16 + row][kb + 16];
            af[mi] = __builtin_shufflevector(lo, hi,
                0,1,2,3,4,5,6,7,8,9,10,11,12,13,14,15);
        }
#pragma unroll
        for (int ni = 0; ni < 4; ++ni) {
            v8bf lo = *(const v8bf*)&Bs[cur][wn + ni * 16 + row][kb];
            v8bf hi = *(const v8bf*)&Bs[cur][wn + ni * 16 + row][kb + 16];
            bfr[ni] = __builtin_shufflevector(lo, hi,
                0,1,2,3,4,5,6,7,8,9,10,11,12,13,14,15);
        }
#pragma unroll
        for (int mi = 0; mi < 2; ++mi)
#pragma unroll
            for (int ni = 0; ni < 4; ++ni)
                acc[mi][ni] = __builtin_amdgcn_wmma_f32_16x16x32_bf16(
                    false, af[mi], false, bfr[ni],
                    (short)0, acc[mi][ni], false, false);

        // ---- write staged registers into buffer[nxt] -----------------------
        if (havenext) {
#pragma unroll
            for (int i = 0; i < 2; ++i)
                *(v8bf*)&As[nxt][sm[i]][sk[i]] = nA[i];
            if (TRANSB) {
#pragma unroll
                for (int i = 0; i < 2; ++i)
                    *(v8bf*)&Bs[nxt][sm[i]][sk[i]] = nB[i];
            } else {
#pragma unroll
                for (int i = 0; i < 2; ++i)
#pragma unroll
                    for (int j = 0; j < 8; ++j)
                        Bs[nxt][xn[i] + j][xk[i]] = nB[i][j];
            }
        }
        cur = nxt;
    }

    // ---- store: C/D layout VGPR r, lanes0-15 M=r, lanes16-31 M=8+r ---------
    // Per-row params hoisted out of the ni loop; no runtime branches.
    const int colOff  = lane & 15;
    const int rowHalf = (lane >> 4) * 8;
#pragma unroll
    for (int mi = 0; mi < 2; ++mi) {
#pragma unroll
        for (int r = 0; r < 8; ++r) {
            const int gr = m0 + wm + mi * 16 + rowHalf + r;
            float badd = 0.f, inv = 0.f, mean = 0.f, bet = 0.f;
            if (HAS_BIAS) badd = bias[gr];
            if (MODE == 2) {
                inv  = gamma[gr] * rsqrtf(rvar[gr] + 1e-5f);
                mean = rmean[gr];
                bet  = beta[gr];
            }
#pragma unroll
            for (int ni = 0; ni < 4; ++ni) {
                const int gc = n0 + wn + ni * 16 + colOff;
                float v = acc[mi][ni][r] + badd;
                size_t idx = (size_t)b * sD + (size_t)gr * CDIM + gc;
                if (MODE == 0) {
                    ((unsigned short*)D)[idx] = f2bf(v);
                } else if (MODE == 1) {
                    ((float*)D)[idx] = v;
                } else {
                    v = v >= 0.f ? v : 0.01f * v;                    // LeakyReLU
                    v = (v - mean) * inv + bet;                      // BN eval
                    v += xres[(size_t)b * MATEL + (size_t)gr * CDIM + gc]; // +x
                    ((float*)D)[idx] = v;
                }
            }
        }
    }
}

// ---------------------------------------------------------------------------
// Host-side orchestration
// ---------------------------------------------------------------------------
extern "C" void kernel_launch(void* const* d_in, const int* in_sizes, int n_in,
                              void* d_out, int out_size, void* d_ws, size_t ws_size,
                              hipStream_t stream) {
    (void)in_sizes; (void)n_in; (void)out_size; (void)ws_size;

    const float* x     = (const float*)d_in[0];
    const float* Wf    = (const float*)d_in[1];
    const float* bf    = (const float*)d_in[2];
    const float* Wg    = (const float*)d_in[3];
    const float* bg    = (const float*)d_in[4];
    const float* Wh    = (const float*)d_in[5];
    const float* bh    = (const float*)d_in[6];
    const float* Wc    = (const float*)d_in[7];
    const float* bc    = (const float*)d_in[8];
    const float* gamma = (const float*)d_in[9];
    const float* beta  = (const float*)d_in[10];
    const float* rmean = (const float*)d_in[11];
    const float* rvar  = (const float*)d_in[12];

    // workspace carving (all regions 256B-aligned)
    char* base = (char*)d_ws;
    size_t off = 0;
    auto carve = [&](size_t bytes) -> void* {
        void* r = base + off;
        off += (bytes + 255) & ~(size_t)255;
        return r;
    };
    __bf16* Xb   = (__bf16*)carve(BATCH * MATEL * 2);   // x, bf16
    __bf16* Wfb  = (__bf16*)carve(MATEL * 2);
    __bf16* Wgb  = (__bf16*)carve(MATEL * 2);
    __bf16* Whb  = (__bf16*)carve(MATEL * 2);
    __bf16* Wcb  = (__bf16*)carve(MATEL * 2);
    __bf16* F    = (__bf16*)carve(BATCH * MATEL * 2);
    __bf16* G    = (__bf16*)carve(BATCH * MATEL * 2);
    __bf16* Hh   = (__bf16*)carve(BATCH * MATEL * 2);
    float*  S    = (float*)carve(BATCH * MATEL * 4);    // scores fp32
    __bf16* Attn = (__bf16*)carve(BATCH * MATEL * 2);
    __bf16* HFG  = (__bf16*)carve(BATCH * MATEL * 2);

    // 1) fp32 -> bf16 conversions
    cvt_f32_bf16<<<(BATCH * MATEL) / 1024, 256, 0, stream>>>(
        x, (unsigned short*)Xb, (int)(BATCH * MATEL));
    cvt_f32_bf16<<<MATEL / 1024, 256, 0, stream>>>(Wf, (unsigned short*)Wfb, (int)MATEL);
    cvt_f32_bf16<<<MATEL / 1024, 256, 0, stream>>>(Wg, (unsigned short*)Wgb, (int)MATEL);
    cvt_f32_bf16<<<MATEL / 1024, 256, 0, stream>>>(Wh, (unsigned short*)Whb, (int)MATEL);
    cvt_f32_bf16<<<MATEL / 1024, 256, 0, stream>>>(Wc, (unsigned short*)Wcb, (int)MATEL);

    dim3 g(8, 8, BATCH), blk(256);

    // 2) projections: F/G/H = W* @ X + b*   (bf16 out)
    gemm_wmma<0, false, true><<<g, blk, 0, stream>>>(Wfb, 0, Xb, (long long)MATEL,
        F, (long long)MATEL, bf, nullptr, nullptr, nullptr, nullptr, nullptr);
    gemm_wmma<0, false, true><<<g, blk, 0, stream>>>(Wgb, 0, Xb, (long long)MATEL,
        G, (long long)MATEL, bg, nullptr, nullptr, nullptr, nullptr, nullptr);
    gemm_wmma<0, false, true><<<g, blk, 0, stream>>>(Whb, 0, Xb, (long long)MATEL,
        Hh, (long long)MATEL, bh, nullptr, nullptr, nullptr, nullptr, nullptr);

    // 3) scores = F @ G^T  (fp32 out)
    gemm_wmma<1, true, false><<<g, blk, 0, stream>>>(F, (long long)MATEL, G, (long long)MATEL,
        S, (long long)MATEL, nullptr, nullptr, nullptr, nullptr, nullptr, nullptr);

    // 4) column softmax -> Attn (bf16)
    softmax_col_kernel<<<dim3(CDIM / 256, BATCH), 256, 0, stream>>>(
        S, (unsigned short*)Attn);

    // 5) HFG = Hh @ Attn  (bf16 out, no bias)
    gemm_wmma<0, false, false><<<g, blk, 0, stream>>>(Hh, (long long)MATEL, Attn, (long long)MATEL,
        HFG, (long long)MATEL, nullptr, nullptr, nullptr, nullptr, nullptr, nullptr);

    // 6) out = BN(LeakyReLU(Wc @ HFG + bc)) + x  (fp32 out, fused epilogue)
    gemm_wmma<2, false, true><<<g, blk, 0, stream>>>(Wcb, 0, HFG, (long long)MATEL,
        d_out, (long long)MATEL, bc, gamma, beta, rmean, rvar, x);
}
